// ACMix_85091892068613
// MI455X (gfx1250) — compile-verified, matrix operand order checked by reading, not run on hardware
//
#include <hip/hip_runtime.h>
#include <hip/hip_bf16.h>

// ---------------------------------------------------------------------------
// ACMix for MI455X (gfx1250): all stages on v_wmma_f32_16x16x32_bf16.
// Shapes: B=8, C_in=256, H=W=32 (N=1024), C_out=256, heads=8, hd=32.
// GEMM blocks: 128x128 tile, 8 waves, each wave 32x64 (2x4 WMMA frags).
// Attention: flash-style, 32 query rows/wave, K/V staged with
// global_load_async_to_lds_b128 (ASYNCcnt).
// ---------------------------------------------------------------------------

typedef __bf16 bf16_t;
typedef __attribute__((ext_vector_type(16))) __bf16 v16bf;
typedef __attribute__((ext_vector_type(8)))  __bf16 v8bf;
typedef __attribute__((ext_vector_type(8)))  float  v8f;

#define NCOL 8192
#define QKV_BYTES (3u * 8u * 8u * 1024u * 32u * 2u)   // 12,582,912
#define MAT_BYTES (256u * 8192u * 4u)                  // 8,388,608

__device__ __forceinline__ bf16_t f2bf(float f) { return (bf16_t)f; }

// 16x32 bf16 fragment (A operand / B^T view). ISA layout:
// lanes 0-15: row=lane, K=0..7 then 16..23; lanes 16-31: row=lane-16,
// K=8..15 then 24..31. Element (r,k) at p[r*rs + k*ks].
__device__ __forceinline__ v16bf load_frag(const bf16_t* p, int rs, int ks) {
  const int lane = threadIdx.x & 31;
  const int r  = lane & 15;
  const int kb = (lane >> 4) << 3;
  v16bf f;
#pragma unroll
  for (int j = 0; j < 8; ++j) f[j]     = p[r * rs + (kb + j) * ks];
#pragma unroll
  for (int j = 0; j < 8; ++j) f[j + 8] = p[r * rs + (kb + 16 + j) * ks];
  return f;
}

__device__ __forceinline__ v8f wmma_bf16(v16bf a, v16bf b, v8f c) {
  return __builtin_amdgcn_wmma_f32_16x16x32_bf16(false, a, false, b,
                                                 (short)0, c, false, false);
}

// CDNA5 async copy: 16 bytes global -> LDS, tracked by ASYNCcnt.
__device__ __forceinline__ void async_b128(bf16_t* lds_dst, const bf16_t* gsrc) {
  unsigned dst = (unsigned)(unsigned long long)lds_dst;  // LDS byte offset
  asm volatile("global_load_async_to_lds_b128 %0, %1, off"
               :: "v"(dst), "v"(gsrc) : "memory");
}
__device__ __forceinline__ void wait_async0() {
  asm volatile("s_wait_asynccnt 0x0" ::: "memory");
}

// ---------------------------------------------------------------------------
// Kernel 1: qkv = shift_w (768x256) @ X (256x8192) + shift_b -> bf16
// layout qkv[which][b][h][n][d].
// ---------------------------------------------------------------------------
__global__ __launch_bounds__(256) void qkv_gemm(const float* __restrict__ x,
                                                const float* __restrict__ shift_w,
                                                const float* __restrict__ shift_b,
                                                bf16_t* __restrict__ qkv) {
  __shared__ bf16_t As[128 * 32];
  __shared__ bf16_t Bs[128 * 32];
  const int t    = threadIdx.x;
  const int wave = t >> 5;
  const int wm   = wave & 3;      // 4 row-groups of 32
  const int wn   = wave >> 2;     // 2 col-groups of 64
  const int m0   = blockIdx.y * 128;
  const int n0   = blockIdx.x * 128;

  v8f zero = {};
  v8f acc[2][4] = {{zero, zero, zero, zero}, {zero, zero, zero, zero}};

  for (int k0 = 0; k0 < 256; k0 += 32) {
    __syncthreads();
#pragma unroll
    for (int i = 0; i < 16; ++i) {                       // A tile 128x32
      int idx = t + i * 256;
      int row = idx >> 5, kk = idx & 31;
      As[idx] = f2bf(shift_w[(m0 + row) * 256 + k0 + kk]);
    }
#pragma unroll
    for (int i = 0; i < 16; ++i) {                       // B^T tile 128x32
      int idx = t + i * 256;
      int nn = idx >> 5, kk = idx & 31;
      int col = n0 + nn;
      int b = col >> 10, hw = col & 1023;
      Bs[idx] = f2bf(x[(b * 256 + (k0 + kk)) * 1024 + hw]);
    }
    if (k0 + 32 < 256) {                                 // global_prefetch_b8
      __builtin_prefetch(&shift_w[(m0 + (t >> 1)) * 256 + k0 + 32], 0, 0);
      __builtin_prefetch(&x[(((n0 >> 10) * 256 + k0 + 32 + (t & 31)) * 1024)], 0, 0);
    }
    __syncthreads();
    v16bf a0 = load_frag(&As[(wm * 32)      * 32], 32, 1);
    v16bf a1 = load_frag(&As[(wm * 32 + 16) * 32], 32, 1);
#pragma unroll
    for (int fn = 0; fn < 4; ++fn) {
      v16bf bf = load_frag(&Bs[(wn * 64 + fn * 16) * 32], 32, 1);
      acc[0][fn] = wmma_bf16(a0, bf, acc[0][fn]);
      acc[1][fn] = wmma_bf16(a1, bf, acc[1][fn]);
    }
  }

  const int lane = t & 31;
  const int ln   = lane & 15;
  const int mg   = (lane >> 4) * 8;
#pragma unroll
  for (int fm = 0; fm < 2; ++fm)
#pragma unroll
    for (int fn = 0; fn < 4; ++fn) {
      int ob  = m0 + wm * 32 + fm * 16 + mg;             // 8 consecutive chans
      int col = n0 + wn * 64 + fn * 16 + ln;
      int which = ob >> 8, h = (ob >> 5) & 7, d0 = ob & 31;
      int b = col >> 10, n = col & 1023;
      v8bf pk;
#pragma unroll
      for (int g = 0; g < 8; ++g) pk[g] = f2bf(acc[fm][fn][g] + shift_b[ob + g]);
      *(v8bf*)&qkv[((((which * 8 + b) * 8 + h) * 1024 + n) << 5) + d0] = pk;
    }
}

// ---------------------------------------------------------------------------
// Kernel 2: flash attention. Block = 256 query rows x one (b,h); each wave
// owns 32 query rows (2 Q frags). K/V tiles staged via async LDS copies.
// Output fp32 matrix attn_out[c][col], c = h*32+d, col = b*1024+n.
// ---------------------------------------------------------------------------
__global__ __launch_bounds__(256) void attn_kernel(const bf16_t* __restrict__ qkv,
                                                   float* __restrict__ attn_out) {
  __shared__ bf16_t Ks[32 * 32];
  __shared__ bf16_t Vs[32 * 32];
  __shared__ bf16_t Ps[8][32 * 32];
  const int t    = threadIdx.x;
  const int wave = t >> 5;
  const int lane = t & 31;
  const int bh   = blockIdx.y;            // 0..63
  const int bb   = bh >> 3, h = bh & 7;
  const int q0   = blockIdx.x * 256 + wave * 32;

  const bf16_t* qb = qkv + (size_t)(((0 * 8 + bb) * 8 + h) * 1024) * 32;
  const bf16_t* kb = qkv + (size_t)(((1 * 8 + bb) * 8 + h) * 1024) * 32;
  const bf16_t* vb = qkv + (size_t)(((2 * 8 + bb) * 8 + h) * 1024) * 32;

  v16bf qf0 = load_frag(qb + q0 * 32,        32, 1);
  v16bf qf1 = load_frag(qb + (q0 + 16) * 32, 32, 1);

  v8f o[2][2] = {{v8f{}, v8f{}}, {v8f{}, v8f{}}};
  float mrun[2][8], lrun[2][8];
#pragma unroll
  for (int qi = 0; qi < 2; ++qi)
#pragma unroll
    for (int g = 0; g < 8; ++g) { mrun[qi][g] = -3.0e38f; lrun[qi][g] = 0.f; }
  const float scale = 0.17677669529663687f;   // 1/sqrt(32)

  for (int kt = 0; kt < 1024; kt += 32) {
    __syncthreads();                           // prior tile fully consumed
    {                                          // async stage K (t<128) / V
      const int tt = t & 127;                  // 128 x 16B chunks per tile
      const bf16_t* src = ((t < 128) ? kb : vb) + (size_t)kt * 32 + tt * 8;
      bf16_t* dst = (t < 128) ? &Ks[tt * 8] : &Vs[tt * 8];
      async_b128(dst, src);
    }
    wait_async0();
    __syncthreads();

    v16bf kf0 = load_frag(&Ks[0],       32, 1);
    v16bf kf1 = load_frag(&Ks[16 * 32], 32, 1);
    v8f s[2][2] = {{v8f{}, v8f{}}, {v8f{}, v8f{}}};
    s[0][0] = wmma_bf16(qf0, kf0, s[0][0]);
    s[0][1] = wmma_bf16(qf0, kf1, s[0][1]);
    s[1][0] = wmma_bf16(qf1, kf0, s[1][0]);
    s[1][1] = wmma_bf16(qf1, kf1, s[1][1]);

    const int pr = (lane >> 4) * 8;
    const int pc = lane & 15;
    bf16_t* pw = &Ps[wave][0];
#pragma unroll
    for (int qi = 0; qi < 2; ++qi) {
#pragma unroll
      for (int g = 0; g < 8; ++g) {
        s[qi][0][g] *= scale; s[qi][1][g] *= scale;
        float m = fmaxf(s[qi][0][g], s[qi][1][g]);
#pragma unroll
        for (int off = 8; off >= 1; off >>= 1)  // stays in 16-lane half
          m = fmaxf(m, __shfl_xor(m, off, 32));
        float mn = fmaxf(mrun[qi][g], m);
        float corr = __expf(mrun[qi][g] - mn);
        mrun[qi][g] = mn;
        float p0 = __expf(s[qi][0][g] - mn);
        float p1 = __expf(s[qi][1][g] - mn);
        float rs = p0 + p1;
#pragma unroll
        for (int off = 8; off >= 1; off >>= 1)
          rs += __shfl_xor(rs, off, 32);
        lrun[qi][g] = lrun[qi][g] * corr + rs;
        o[qi][0][g] *= corr; o[qi][1][g] *= corr;
        pw[(qi * 16 + pr + g) * 32 + pc]      = f2bf(p0);
        pw[(qi * 16 + pr + g) * 32 + pc + 16] = f2bf(p1);
      }
    }
    v16bf vf0 = load_frag(&Vs[0],  1, 32);      // V cols d=0..15 (B^T view)
    v16bf vf1 = load_frag(&Vs[16], 1, 32);      // V cols d=16..31
#pragma unroll
    for (int qi = 0; qi < 2; ++qi) {
      v16bf pf = load_frag(pw + qi * 16 * 32, 32, 1);
      o[qi][0] = wmma_bf16(pf, vf0, o[qi][0]);
      o[qi][1] = wmma_bf16(pf, vf1, o[qi][1]);
    }
  }

  const int dloc = lane & 15;
  const int mg   = (lane >> 4) * 8;
#pragma unroll
  for (int qi = 0; qi < 2; ++qi)
#pragma unroll
    for (int g = 0; g < 8; ++g) {
      float inv = 1.0f / lrun[qi][g];
      int q  = q0 + qi * 16 + mg + g;
      int c0 = h * 32 + dloc;
      attn_out[c0 * NCOL + bb * 1024 + q]        = o[qi][0][g] * inv;
      attn_out[(c0 + 16) * NCOL + bb * 1024 + q] = o[qi][1][g] * inv;
    }
}

// ---------------------------------------------------------------------------
// Kernel 3: 3x3 conv as implicit GEMM: conv_w (256x2304) @ im2col(x).
// ---------------------------------------------------------------------------
__global__ __launch_bounds__(256) void conv_gemm(const float* __restrict__ x,
                                                 const float* __restrict__ conv_w,
                                                 float* __restrict__ conv_out) {
  __shared__ bf16_t As[128 * 32];
  __shared__ bf16_t Bs[128 * 32];
  const int t    = threadIdx.x;
  const int wave = t >> 5;
  const int wm   = wave & 3;
  const int wn   = wave >> 2;
  const int m0   = blockIdx.y * 128;
  const int n0   = blockIdx.x * 128;

  v8f zero = {};
  v8f acc[2][4] = {{zero, zero, zero, zero}, {zero, zero, zero, zero}};

  for (int k0 = 0; k0 < 2304; k0 += 32) {
    __syncthreads();
#pragma unroll
    for (int i = 0; i < 16; ++i) {                       // A: conv_w row-major
      int idx = t + i * 256;
      int row = idx >> 5, kk = idx & 31;
      As[idx] = f2bf(conv_w[(m0 + row) * 2304 + k0 + kk]);
    }
#pragma unroll
    for (int i = 0; i < 16; ++i) {                       // B: im2col gather
      int idx = t + i * 256;
      int nn = idx >> 5, kk = idx & 31;
      int k  = k0 + kk;
      int c  = k / 9, r9 = k - c * 9;
      int kh = r9 / 3, kw = r9 - kh * 3;
      int col = n0 + nn;
      int b = col >> 10, hw = col & 1023;
      int y = (hw >> 5) + kh - 1, xx = (hw & 31) + kw - 1;
      float v = 0.f;
      if ((unsigned)y < 32u && (unsigned)xx < 32u)
        v = x[(b * 256 + c) * 1024 + y * 32 + xx];
      Bs[idx] = f2bf(v);
    }
    if (k0 + 32 < 2304)
      __builtin_prefetch(&conv_w[(m0 + (t >> 1)) * 2304 + k0 + 32], 0, 0);
    __syncthreads();
    v16bf a0 = load_frag(&As[(wm * 32)      * 32], 32, 1);
    v16bf a1 = load_frag(&As[(wm * 32 + 16) * 32], 32, 1);
#pragma unroll
    for (int fn = 0; fn < 4; ++fn) {
      v16bf bf = load_frag(&Bs[(wn * 64 + fn * 16) * 32], 32, 1);
      acc[0][fn] = wmma_bf16(a0, bf, acc[0][fn]);
      acc[1][fn] = wmma_bf16(a1, bf, acc[1][fn]);
    }
  }

  const int lane = t & 31;
  const int ln   = lane & 15;
  const int mg   = (lane >> 4) * 8;
#pragma unroll
  for (int fm = 0; fm < 2; ++fm)
#pragma unroll
    for (int fn = 0; fn < 4; ++fn)
#pragma unroll
      for (int g = 0; g < 8; ++g) {
        int o   = m0 + wm * 32 + fm * 16 + mg + g;
        int col = n0 + wn * 64 + fn * 16 + ln;
        conv_out[o * NCOL + col] = acc[fm][fn][g];
      }
}

// ---------------------------------------------------------------------------
// Kernel 4: mixed = bf16( alpha*(attn+pos) + (1-alpha)*conv )
// ---------------------------------------------------------------------------
__global__ __launch_bounds__(256) void blend_kernel(const float* __restrict__ attn,
                                                    const float* __restrict__ conv,
                                                    const float* __restrict__ pos,
                                                    const float* __restrict__ alpha,
                                                    bf16_t* __restrict__ mixed) {
  int idx = blockIdx.x * 256 + threadIdx.x;   // 256*8192 exactly
  float a = alpha[0];
  int c = idx >> 13;
  mixed[idx] = f2bf(a * (attn[idx] + pos[c]) + (1.f - a) * conv[idx]);
}

// ---------------------------------------------------------------------------
// Kernel 5: out = proj_w (256x256) @ mixed (256x8192) + proj_b -> fp32 NCHW
// ---------------------------------------------------------------------------
__global__ __launch_bounds__(256) void proj_gemm(const bf16_t* __restrict__ mixed,
                                                 const float* __restrict__ proj_w,
                                                 const float* __restrict__ proj_b,
                                                 float* __restrict__ out) {
  __shared__ bf16_t As[128 * 32];
  __shared__ bf16_t Bs[128 * 32];
  const int t    = threadIdx.x;
  const int wave = t >> 5;
  const int wm   = wave & 3;
  const int wn   = wave >> 2;
  const int m0   = blockIdx.y * 128;
  const int n0   = blockIdx.x * 128;

  v8f zero = {};
  v8f acc[2][4] = {{zero, zero, zero, zero}, {zero, zero, zero, zero}};

  for (int k0 = 0; k0 < 256; k0 += 32) {
    __syncthreads();
#pragma unroll
    for (int i = 0; i < 16; ++i) {
      int idx = t + i * 256;
      int row = idx >> 5, kk = idx & 31;
      As[idx] = f2bf(proj_w[(m0 + row) * 256 + k0 + kk]);
    }
#pragma unroll
    for (int i = 0; i < 16; ++i) {               // mixed already bf16
      int idx = t + i * 256;
      int nn = idx >> 5, kk = idx & 31;
      Bs[idx] = mixed[(k0 + kk) * NCOL + n0 + nn];
    }
    if (k0 + 32 < 256)
      __builtin_prefetch(&mixed[(k0 + 32 + (t & 31)) * NCOL + n0], 0, 0);
    __syncthreads();
    v16bf a0 = load_frag(&As[(wm * 32)      * 32], 32, 1);
    v16bf a1 = load_frag(&As[(wm * 32 + 16) * 32], 32, 1);
#pragma unroll
    for (int fn = 0; fn < 4; ++fn) {
      v16bf bf = load_frag(&Bs[(wn * 64 + fn * 16) * 32], 32, 1);
      acc[0][fn] = wmma_bf16(a0, bf, acc[0][fn]);
      acc[1][fn] = wmma_bf16(a1, bf, acc[1][fn]);
    }
  }

  const int lane = t & 31;
  const int ln   = lane & 15;
  const int mg   = (lane >> 4) * 8;
#pragma unroll
  for (int fm = 0; fm < 2; ++fm)
#pragma unroll
    for (int fn = 0; fn < 4; ++fn)
#pragma unroll
      for (int g = 0; g < 8; ++g) {
        int o   = m0 + wm * 32 + fm * 16 + mg + g;
        int col = n0 + wn * 64 + fn * 16 + ln;
        int b = col >> 10, hw = col & 1023;
        out[(b * 256 + o) * 1024 + hw] = acc[fm][fn][g] + proj_b[o];
      }
}

// ---------------------------------------------------------------------------
extern "C" void kernel_launch(void* const* d_in, const int* in_sizes, int n_in,
                              void* d_out, int out_size, void* d_ws, size_t ws_size,
                              hipStream_t stream) {
  const float* x       = (const float*)d_in[0];
  const float* shift_w = (const float*)d_in[1];
  const float* shift_b = (const float*)d_in[2];
  const float* conv_w  = (const float*)d_in[3];
  const float* pos     = (const float*)d_in[4];
  const float* proj_w  = (const float*)d_in[5];
  const float* proj_b  = (const float*)d_in[6];
  const float* alpha   = (const float*)d_in[7];
  float* out = (float*)d_out;

  char* ws = (char*)d_ws;
  bf16_t* qkv   = (bf16_t*)(ws);
  float*  attn  = (float*)(ws + QKV_BYTES);
  float*  convb = (float*)(ws + QKV_BYTES + MAT_BYTES);
  bf16_t* mixed = (bf16_t*)(ws + QKV_BYTES + 2 * MAT_BYTES);
  (void)in_sizes; (void)n_in; (void)out_size; (void)ws_size;

  qkv_gemm   <<<dim3(64, 6), 256, 0, stream>>>(x, shift_w, shift_b, qkv);
  attn_kernel<<<dim3(4, 64), 256, 0, stream>>>(qkv, attn);
  conv_gemm  <<<dim3(64, 2), 256, 0, stream>>>(x, conv_w, convb);
  blend_kernel<<<8192, 256, 0, stream>>>(attn, convb, pos, alpha, mixed);
  proj_gemm  <<<dim3(64, 2), 256, 0, stream>>>(mixed, proj_w, proj_b, out);
}
